// GPT_12429635355013
// MI455X (gfx1250) — matrix operand order repaired and checked by reading, not compiled, not run on hardware
//
#include <hip/hip_runtime.h>
#include <math.h>

// ---------------- problem constants ----------------
#define VV   50257
#define DD   1024
#define HH   16
#define LL   8
#define TT   1024
#define BB   2
#define DHH  64
#define DFF  4096
#define MM   (BB * TT)      // 2048 rows of activations

typedef unsigned short ushort_t;
typedef __attribute__((ext_vector_type(16))) __bf16 v16bf;
typedef __attribute__((ext_vector_type(8)))  float  v8f;
typedef __attribute__((ext_vector_type(4)))  int    int4v;

#define GLOBAL_AS __attribute__((address_space(1)))
#define LDS_AS    __attribute__((address_space(3)))

#if defined(__has_builtin)
#if __has_builtin(__builtin_amdgcn_global_load_async_to_lds_b128)
#define HAVE_ASYNC 1
#endif
#endif
#ifndef HAVE_ASYNC
#define HAVE_ASYNC 0
#endif

#if HAVE_ASYNC
#if defined(__has_builtin) && __has_builtin(__builtin_amdgcn_s_wait_asynccnt)
#define ASYNC_WAIT() __builtin_amdgcn_s_wait_asynccnt(0)
#else
#define ASYNC_WAIT() asm volatile("s_wait_asynccnt 0x0" ::: "memory")
#endif
#else
#define ASYNC_WAIT()
#endif

union FragBF { v16bf v; uint4 q[2]; };
union Pack8  { ushort_t u[8]; uint4 q; };

__device__ __forceinline__ ushort_t f32_to_bf16(float f) {
    union { float f; unsigned int u; } v; v.f = f;
    unsigned int u = v.u;
    unsigned int r = u + 0x7FFFu + ((u >> 16) & 1u);  // round-to-nearest-even
    return (ushort_t)(r >> 16);
}

// ---------------- embedding: x = tok_emb[idx] + pos_emb ----------------
__global__ void embed_kernel(const int* __restrict__ idx,
                             const float* __restrict__ tok_emb,
                             const float* __restrict__ pos_emb,
                             float* __restrict__ x) {
    int row = blockIdx.x;            // 0..M-1
    int t   = row % TT;
    int tok = idx[row];
    int c   = threadIdx.x * 4;       // 256 threads * 4 = 1024
    const float4 te = *(const float4*)(tok_emb + (size_t)tok * DD + c);
    const float4 pe = *(const float4*)(pos_emb + (size_t)t * DD + c);
    float4 o; o.x = te.x + pe.x; o.y = te.y + pe.y; o.z = te.z + pe.z; o.w = te.w + pe.w;
    *(float4*)(x + (size_t)row * DD + c) = o;
}

// ---------------- layernorm (f32 in -> bf16 out) ----------------
__global__ void layernorm_kernel(const float* __restrict__ x,
                                 const float* __restrict__ w,
                                 const float* __restrict__ b,
                                 ushort_t* __restrict__ out) {
    __shared__ float red[256];
    __shared__ float red2[256];
    int row = blockIdx.x;
    int tid = threadIdx.x;
    const float* xr = x + (size_t)row * DD;
    float4 v = *(const float4*)(xr + tid * 4);
    float s  = v.x + v.y + v.z + v.w;
    float s2 = v.x * v.x + v.y * v.y + v.z * v.z + v.w * v.w;
    red[tid] = s; red2[tid] = s2;
    __syncthreads();
    for (int off = 128; off > 0; off >>= 1) {
        if (tid < off) { red[tid] += red[tid + off]; red2[tid] += red2[tid + off]; }
        __syncthreads();
    }
    float mu   = red[0] * (1.0f / DD);
    float var  = red2[0] * (1.0f / DD) - mu * mu;
    float rstd = rsqrtf(var + 1e-5f);
    int c = tid * 4;
    ushort_t* orow = out + (size_t)row * DD;
    #pragma unroll
    for (int i = 0; i < 4; i++) {
        float xv = ((const float*)&v)[i];
        float hv = (xv - mu) * rstd * w[c + i] + b[c + i];
        orow[c + i] = f32_to_bf16(hv);
    }
}

// ---------------- GEMM tile staging helpers ----------------
// A tile: 128 rows x 32 k of bf16 = 512 16B chunks; 256 threads -> 2 chunks each.
struct AStage { uint4 a0, a1; };

__device__ __forceinline__ void stageA_issue(const ushort_t* __restrict__ A, int K, int gm, int k0,
                                             ushort_t (*sA)[32], AStage& st) {
    const int t  = threadIdx.x;
    const int c0 = t, c1 = t + 256;
    const ushort_t* g0 = A + (size_t)(gm + (c0 >> 2)) * K + k0 + (c0 & 3) * 8;
    const ushort_t* g1 = A + (size_t)(gm + (c1 >> 2)) * K + k0 + (c1 & 3) * 8;
#if HAVE_ASYNC
    __builtin_amdgcn_global_load_async_to_lds_b128(
        (GLOBAL_AS int4v*)((GLOBAL_AS const void*)g0),
        (LDS_AS int4v*)((LDS_AS void*)&sA[c0 >> 2][(c0 & 3) * 8]), 0, 0);
    __builtin_amdgcn_global_load_async_to_lds_b128(
        (GLOBAL_AS int4v*)((GLOBAL_AS const void*)g1),
        (LDS_AS int4v*)((LDS_AS void*)&sA[c1 >> 2][(c1 & 3) * 8]), 0, 0);
    (void)st;
#else
    st.a0 = *(const uint4*)g0;
    st.a1 = *(const uint4*)g1;
#endif
}

__device__ __forceinline__ void stageA_commit(ushort_t (*sA)[32], const AStage& st) {
#if !HAVE_ASYNC
    const int t = threadIdx.x;
    *(uint4*)&sA[t >> 2][(t & 3) * 8]                 = st.a0;
    *(uint4*)&sA[(t + 256) >> 2][((t + 256) & 3) * 8] = st.a1;
#else
    (void)sA; (void)st;
#endif
}

// B tile: 64 rows x 32 k of f32 -> bf16; 256 threads -> 8 floats each.
__device__ __forceinline__ void loadB_issue(const float* __restrict__ W, int K, int gn, int k0,
                                            int Nact, float4& f0, float4& f1) {
    const int t = threadIdx.x;
    const int row = t >> 2, seg = t & 3;
    const int nrow = gn + row;
    if (nrow < Nact) {
        const float* src = W + (size_t)nrow * K + k0 + seg * 8;
        f0 = *(const float4*)(src);
        f1 = *(const float4*)(src + 4);
        if (k0 + 32 < K) __builtin_prefetch(src + 32, 0, 1);
    } else {
        f0.x = f0.y = f0.z = f0.w = 0.0f;
        f1 = f0;
    }
}

__device__ __forceinline__ void storeB(ushort_t (*sB)[32], const float4& f0, const float4& f1) {
    const int t = threadIdx.x;
    const int row = t >> 2, seg = t & 3;
    Pack8 p;
    p.u[0] = f32_to_bf16(f0.x); p.u[1] = f32_to_bf16(f0.y);
    p.u[2] = f32_to_bf16(f0.z); p.u[3] = f32_to_bf16(f0.w);
    p.u[4] = f32_to_bf16(f1.x); p.u[5] = f32_to_bf16(f1.y);
    p.u[6] = f32_to_bf16(f1.z); p.u[7] = f32_to_bf16(f1.w);
    *(uint4*)&sB[row][seg * 8] = p.q;
}

// ---------------- WMMA GEMM:  C[M,N] = A[M,K](bf16) * W[N,K]^T(f32) ----------------
// Block tile 128(M) x 64(N), BK=32, 256 threads = 8 waves, wave -> 32x32, double-buffered LDS.
// grid.x = M/128 (fast: consecutive blocks reuse the same weight panel through L2)
// grid.y = ceil(N/64)
// EPI 0: outF = acc + bias              (qkv, lm_head)
// EPI 1: outF += acc + bias             (residual add into x)
// EPI 2: outBF = bf16(gelu(acc + bias)) (FF1)
template <int EPI>
__global__ __launch_bounds__(256)
void gemm_bf16_wmma(const ushort_t* __restrict__ A,
                    const float* __restrict__ W,
                    const float* __restrict__ bias,
                    float* __restrict__ outF,
                    ushort_t* __restrict__ outBF,
                    int K, int Nact) {
    __shared__ ushort_t sA[2][128][32];
    __shared__ ushort_t sB[2][64][32];

    const int tid  = threadIdx.x;
    const int lane = tid & 31;
    const int wave = tid >> 5;         // 0..7
    const int wm   = wave >> 1;        // 0..3  (32-row strip)
    const int wn   = wave & 1;         // 0..1  (32-col strip)
    const int gm   = blockIdx.x * 128;
    const int gn   = blockIdx.y * 64;

    v8f c[2][2];
    #pragma unroll
    for (int i = 0; i < 2; i++)
        #pragma unroll
        for (int j = 0; j < 2; j++)
            c[i][j] = (v8f){0.f, 0.f, 0.f, 0.f, 0.f, 0.f, 0.f, 0.f};

    AStage ast;
    float4 bf0, bf1;

    // prologue: fill buffer 0
    stageA_issue(A, K, gm, 0, sA[0], ast);
    loadB_issue(W, K, gn, 0, Nact, bf0, bf1);
    stageA_commit(sA[0], ast);
    storeB(sB[0], bf0, bf1);
    ASYNC_WAIT();
    __syncthreads();

    const int KT = K >> 5;
    for (int kt = 0; kt < KT; ++kt) {
        const int cur  = kt & 1;
        const int nxt  = cur ^ 1;
        const bool more = (kt + 1 < KT);
        if (more) {
            stageA_issue(A, K, gm, (kt + 1) * 32, sA[nxt], ast);
            loadB_issue(W, K, gn, (kt + 1) * 32, Nact, bf0, bf1);
        }

        // ---- fragments per CDNA5 16-bit WMMA layouts ----
        // A 16x32: lane L holds row m=(L&15); lanes 0-15 K:{0..7,16..23}, lanes 16-31 K:{8..15,24..31}
        FragBF fa[2];
        {
            const int m  = wm * 32 + (lane & 15);
            const int kb = (lane < 16) ? 0 : 8;
            #pragma unroll
            for (int mi = 0; mi < 2; mi++) {
                fa[mi].q[0] = *(const uint4*)&sA[cur][m + mi * 16][kb];
                fa[mi].q[1] = *(const uint4*)&sA[cur][m + mi * 16][kb + 16];
            }
        }
        // B 32x16: lane holds col n=(lane&15); lanes 0-15 K:{0..7,16..23}, lanes 16-31 K:{8..15,24..31}
        FragBF fb[2];
        {
            const int n  = wn * 32 + (lane & 15);
            const int kb = (lane < 16) ? 0 : 8;
            #pragma unroll
            for (int ni = 0; ni < 2; ni++) {
                fb[ni].q[0] = *(const uint4*)&sB[cur][n + ni * 16][kb];
                fb[ni].q[1] = *(const uint4*)&sB[cur][n + ni * 16][kb + 16];
            }
        }
        #pragma unroll
        for (int mi = 0; mi < 2; mi++)
            #pragma unroll
            for (int ni = 0; ni < 2; ni++)
                c[mi][ni] = __builtin_amdgcn_wmma_f32_16x16x32_bf16(
                    false, fa[mi].v, false, fb[ni].v, (short)0, c[mi][ni], false, false);

        if (more) {
            stageA_commit(sA[nxt], ast);
            storeB(sB[nxt], bf0, bf1);
            ASYNC_WAIT();
            __syncthreads();
        }
    }

    // ---- epilogue: C/D layout lanes 0-15 -> M=r, lanes 16-31 -> M=8+r, N=lane&15 ----
    #pragma unroll
    for (int mi = 0; mi < 2; mi++) {
        #pragma unroll
        for (int ni = 0; ni < 2; ni++) {
            int col = gn + wn * 32 + ni * 16 + (lane & 15);
            if (col >= Nact) continue;
            float bv = bias ? bias[col] : 0.0f;
            #pragma unroll
            for (int r = 0; r < 8; r++) {
                int row = gm + wm * 32 + mi * 16 + ((lane < 16) ? r : 8 + r);
                float val = c[mi][ni][r] + bv;
                size_t off = (size_t)row * Nact + col;
                if (EPI == 0) {
                    outF[off] = val;
                } else if (EPI == 1) {
                    outF[off] += val;
                } else { // gelu -> bf16
                    float g = 0.5f * val * (1.0f + erff(val * 0.70710678118f));
                    outBF[off] = f32_to_bf16(g);
                }
            }
        }
    }
}

// ---------------- attention: one wave per query row, online softmax ----------------
__global__ __launch_bounds__(256)
void attention_kernel(const float* __restrict__ qkv, ushort_t* __restrict__ o_out) {
    __shared__ float qv[8][64];
    const int lane = threadIdx.x & 31;
    const int w    = threadIdx.x >> 5;
    const int rowId = blockIdx.x * 8 + w;          // 0 .. B*H*T-1
    const int b    = rowId / (HH * TT);
    const int rem  = rowId % (HH * TT);
    const int head = rem / TT;
    const int q    = rem % TT;
    const float scale = 0.125f;                    // 1/sqrt(64)

    const float* qptr = qkv + ((size_t)(b * TT + q) * 3 * DD) + head * DHH;
    qv[w][lane]      = qptr[lane];
    qv[w][lane + 32] = qptr[lane + 32];
    __syncthreads();

    float m = -INFINITY, l = 0.0f;
    float acc0 = 0.0f, acc1 = 0.0f;                // dims lane, lane+32

    for (int k0 = 0; k0 <= q; k0 += 32) {
        int k = k0 + lane;
        float s = -INFINITY;
        if (k <= q) {
            const float* kp = qkv + ((size_t)(b * TT + k) * 3 * DD) + DD + head * DHH;
            float acc = 0.0f;
            #pragma unroll 8
            for (int d = 0; d < DHH; d++) acc += qv[w][d] * kp[d];
            s = acc * scale;
        }
        // wave max / sum
        float smax = s;
        #pragma unroll
        for (int off = 16; off > 0; off >>= 1)
            smax = fmaxf(smax, __shfl_xor(smax, off, 32));
        float mn = fmaxf(m, smax);
        float p  = (k <= q) ? __expf(s - mn) : 0.0f;
        float ps = p;
        #pragma unroll
        for (int off = 16; off > 0; off >>= 1)
            ps += __shfl_xor(ps, off, 32);
        float alpha = __expf(m - mn);
        l = l * alpha + ps;
        m = mn;
        acc0 *= alpha; acc1 *= alpha;
        int kmax = (q - k0 < 31) ? (q - k0) : 31;
        for (int kk = 0; kk <= kmax; kk++) {
            float pv = __shfl(p, kk, 32);
            const float* vp = qkv + ((size_t)(b * TT + k0 + kk) * 3 * DD) + 2 * DD + head * DHH;
            acc0 += pv * vp[lane];
            acc1 += pv * vp[lane + 32];
        }
    }
    float inv = 1.0f / l;
    ushort_t* op = o_out + (size_t)(b * TT + q) * DD + head * DHH;
    op[lane]      = f32_to_bf16(acc0 * inv);
    op[lane + 32] = f32_to_bf16(acc1 * inv);
}

// ---------------- loss ----------------
__global__ void zero_loss_kernel(float* loss) { if (threadIdx.x == 0 && blockIdx.x == 0) *loss = 0.0f; }

__global__ void loss_kernel(const float* __restrict__ logits,
                            const int* __restrict__ targets,
                            float* __restrict__ loss) {
    __shared__ float red[256];
    int row = blockIdx.x;
    int tid = threadIdx.x;
    const float* lr = logits + (size_t)row * VV;
    // pass 1: max
    float mx = -INFINITY;
    for (int c = tid; c < VV; c += 256) mx = fmaxf(mx, lr[c]);
    red[tid] = mx; __syncthreads();
    for (int off = 128; off > 0; off >>= 1) {
        if (tid < off) red[tid] = fmaxf(red[tid], red[tid + off]);
        __syncthreads();
    }
    mx = red[0]; __syncthreads();
    // pass 2: sum exp
    float sm = 0.0f;
    for (int c = tid; c < VV; c += 256) sm += __expf(lr[c] - mx);
    red[tid] = sm; __syncthreads();
    for (int off = 128; off > 0; off >>= 1) {
        if (tid < off) red[tid] += red[tid + off];
        __syncthreads();
    }
    if (tid == 0) {
        float lse = mx + logf(red[0]);
        float lp  = lr[targets[row]] - lse;
        atomicAdd(loss, -lp * (1.0f / (float)MM));
    }
}

// ---------------- host side ----------------
extern "C" void kernel_launch(void* const* d_in, const int* in_sizes, int n_in,
                              void* d_out, int out_size, void* d_ws, size_t ws_size,
                              hipStream_t stream) {
    const int*   idx      = (const int*)d_in[0];
    const int*   targets  = (const int*)d_in[1];
    const float* tok_emb  = (const float*)d_in[2];
    const float* pos_emb  = (const float*)d_in[3];
    const float* lm_head_b= (const float*)d_in[4];
    const float* ln1_w    = (const float*)d_in[5];
    const float* ln1_b    = (const float*)d_in[6];
    const float* qkv_w    = (const float*)d_in[7];
    const float* qkv_b    = (const float*)d_in[8];
    const float* out_w    = (const float*)d_in[9];
    const float* out_b    = (const float*)d_in[10];
    const float* ln2_w    = (const float*)d_in[11];
    const float* ln2_b    = (const float*)d_in[12];
    const float* ff1_w    = (const float*)d_in[13];
    const float* ff1_b    = (const float*)d_in[14];
    const float* ff2_w    = (const float*)d_in[15];
    const float* ff2_b    = (const float*)d_in[16];
    const float* lnf_w    = (const float*)d_in[17];
    const float* lnf_b    = (const float*)d_in[18];

    float* logits = (float*)d_out;                         // [M, V]
    float* loss   = logits + (size_t)MM * VV;              // scalar

    // workspace layout
    float*    xbuf   = (float*)d_ws;                                   // [M, D]  f32
    float*    qkvbuf = xbuf + (size_t)MM * DD;                         // [M, 3D] f32
    ushort_t* hbuf   = (ushort_t*)(qkvbuf + (size_t)MM * 3 * DD);      // [M, D]  bf16
    ushort_t* obuf   = hbuf + (size_t)MM * DD;                         // [M, D]  bf16
    ushort_t* ffbuf  = obuf + (size_t)MM * DD;                         // [M, DF] bf16

    dim3 blk256(256);

    embed_kernel<<<MM, blk256, 0, stream>>>(idx, tok_emb, pos_emb, xbuf);

    for (int l = 0; l < LL; l++) {
        const float* l1w = ln1_w + (size_t)l * DD;
        const float* l1b = ln1_b + (size_t)l * DD;
        const float* qw  = qkv_w + (size_t)l * 3 * DD * DD;
        const float* qb  = qkv_b + (size_t)l * 3 * DD;
        const float* ow  = out_w + (size_t)l * DD * DD;
        const float* ob  = out_b + (size_t)l * DD;
        const float* l2w = ln2_w + (size_t)l * DD;
        const float* l2b = ln2_b + (size_t)l * DD;
        const float* f1w = ff1_w + (size_t)l * DFF * DD;
        const float* f1b = ff1_b + (size_t)l * DFF;
        const float* f2w = ff2_w + (size_t)l * DD * DFF;
        const float* f2b = ff2_b + (size_t)l * DD;

        // ln1 -> h(bf16)
        layernorm_kernel<<<MM, blk256, 0, stream>>>(xbuf, l1w, l1b, hbuf);
        // qkv = h @ qw^T + qb   [M, 3D] f32
        gemm_bf16_wmma<0><<<dim3(MM / 128, 3 * DD / 64), blk256, 0, stream>>>(
            hbuf, qw, qb, qkvbuf, nullptr, DD, 3 * DD);
        // attention -> o(bf16)
        attention_kernel<<<dim3(BB * HH * TT / 8), blk256, 0, stream>>>(qkvbuf, obuf);
        // x += o @ ow^T + ob
        gemm_bf16_wmma<1><<<dim3(MM / 128, DD / 64), blk256, 0, stream>>>(
            obuf, ow, ob, xbuf, nullptr, DD, DD);
        // ln2 -> h(bf16)
        layernorm_kernel<<<MM, blk256, 0, stream>>>(xbuf, l2w, l2b, hbuf);
        // ff = gelu(h @ f1w^T + f1b) -> bf16 [M, DF]
        gemm_bf16_wmma<2><<<dim3(MM / 128, DFF / 64), blk256, 0, stream>>>(
            hbuf, f1w, f1b, nullptr, ffbuf, DD, DFF);
        // x += ff @ f2w^T + f2b
        gemm_bf16_wmma<1><<<dim3(MM / 128, DD / 64), blk256, 0, stream>>>(
            ffbuf, f2w, f2b, xbuf, nullptr, DFF, DD);
    }

    // final layernorm -> h(bf16)
    layernorm_kernel<<<MM, blk256, 0, stream>>>(xbuf, lnf_w, lnf_b, hbuf);
    // logits = h @ tok_emb^T + lm_head_b   [M, V]
    gemm_bf16_wmma<0><<<dim3(MM / 128, (VV + 63) / 64), blk256, 0, stream>>>(
        hbuf, tok_emb, lm_head_b, logits, nullptr, DD, VV);
    // loss
    zero_loss_kernel<<<1, 1, 0, stream>>>(loss);
    loss_kernel<<<MM, blk256, 0, stream>>>(logits, targets, loss);
}
